// NonLocal2dBlock_68367289417927
// MI455X (gfx1250) — matrix operand order repaired and checked by reading, not compiled, not run on hardware
//
#include <hip/hip_runtime.h>
#include <hip/hip_bf16.h>
#include <hip/hip_fp16.h>

// Non-local (self-attention) block for MI455X / gfx1250, wave32 + WMMA f16.
// B=8, C=256, H=W=64 -> N=4096, M=1024, d=32, dv=128.

typedef __attribute__((ext_vector_type(16))) _Float16 v16h;
typedef __attribute__((ext_vector_type(8)))  _Float16 v8h;
typedef __attribute__((ext_vector_type(8)))  float    v8f;

#define C_    256
#define HW_   4096   // 64*64
#define NPIX  4096
#define MPOOL 1024
#define DQ    32
#define DV    128
#define CCAT  192    // 32 theta + 32 phi + 128 g
#define CPG   160    // phi(32) + g(128) full-res conv rows

// ===========================================================================
//                      WMMA-prep pipeline kernels
// ===========================================================================

// x (b,256,4096) f32  ->  xt (b,4096,256) f16   (pixel-major for B fragments)
__global__ void k_transpose_x(const float* __restrict__ x, _Float16* __restrict__ xt) {
  int gid = blockIdx.x * 256 + threadIdx.x;     // b*4096 + i
  int bb = gid >> 12, i = gid & 4095;
  const float* xb = x + (size_t)bb * C_ * HW_ + i;
  _Float16* ob = xt + (size_t)gid * C_;
  for (int cc = 0; cc < 32; ++cc) {
    v8h v;
#pragma unroll
    for (int k = 0; k < 8; ++k) v[k] = (_Float16)xb[(size_t)(cc * 8 + k) * HW_];
    *(v8h*)(ob + cc * 8) = v;
  }
}

// Pack concatenated weights (theta 0-31, phi 32-63, g 64-191; each row 256)
// into WMMA A-fragment order, and build bcat[192] (f32 biases).
// kmap: k = (j>>3)*16 + ((lane>>4)<<3) + (j&7)  within a 32-wide K chunk.
__global__ void k_pack_wcat(const float* __restrict__ tw, const float* __restrict__ pw,
                            const float* __restrict__ gw, const float* __restrict__ tb,
                            const float* __restrict__ pb, const float* __restrict__ gb,
                            _Float16* __restrict__ wpk, float* __restrict__ bcat) {
  int idx = blockIdx.x * 256 + threadIdx.x;     // 49152 total
  int j    = idx & 15;
  int lane = (idx >> 4) & 31;
  int qt   = idx >> 9;                          // 0..95
  int q = qt & 7, mt = qt >> 3;                 // K-chunk, M-tile
  int row = mt * 16 + (lane & 15);
  int k   = q * 32 + ((j >> 3) << 4) + ((lane >> 4) << 3) + (j & 7);
  float v = (row < 32) ? tw[row * C_ + k]
          : (row < 64) ? pw[(row - 32) * C_ + k]
                       : gw[(row - 64) * C_ + k];
  wpk[idx] = (_Float16)v;
  if (idx < CCAT)
    bcat[idx] = (idx < 32) ? tb[idx] : (idx < 64) ? pb[idx - 32] : gb[idx - 64];
}

// Fused theta/phi/g conv via WMMA. One wave per 16-pixel tile.
// B fragments (K=256 of xt) kept in registers across all 12 channel tiles.
// mt 0..1 -> theta, scattered to (b, i, 32) layout; mt 2..11 -> full-res
// conv+bias of phi/g rows into convpg (b, 160, 4096) f16 for later pooling.
__global__ void __launch_bounds__(32)
k_conv_wmma(const _Float16* __restrict__ xt, const _Float16* __restrict__ wpk,
            const float* __restrict__ bcat, _Float16* __restrict__ th,
            _Float16* __restrict__ convpg) {
  const int lane = threadIdx.x;
  const int l15 = lane & 15, hh = lane >> 4;
  const int wid = blockIdx.x;                   // 0..2047
  const int bb = wid >> 8;
  const int i0 = (wid & 255) << 4;

  // ---- B fragments: x columns for this pixel tile, 8 K-chunks of 32
  v16h bf[8];
#pragma unroll
  for (int q = 0; q < 8; ++q) {
    size_t xb = ((size_t)(bb * NPIX + i0 + l15)) * C_ + q * 32 + hh * 8;
    v8h b0 = *(const v8h*)(xt + xb);
    v8h b1 = *(const v8h*)(xt + xb + 16);
#pragma unroll
    for (int j = 0; j < 8; ++j) { bf[q][j] = b0[j]; bf[q][j + 8] = b1[j]; }
  }

  for (int mt = 0; mt < 12; ++mt) {
    v8f acc;
#pragma unroll
    for (int r = 0; r < 8; ++r) acc[r] = 0.f;
#pragma unroll
    for (int q = 0; q < 8; ++q) {
      const _Float16* wb = wpk + ((size_t)((mt * 8 + q) * 32 + lane)) * 16;
      v8h w0 = *(const v8h*)wb;
      v8h w1 = *(const v8h*)(wb + 8);
      v16h aw;
#pragma unroll
      for (int j = 0; j < 8; ++j) { aw[j] = w0[j]; aw[j + 8] = w1[j]; }
      acc = __builtin_amdgcn_wmma_f32_16x16x32_f16(false, aw, false, bf[q],
                                                   (short)0, acc, false, false);
    }
    if (mt < 2) {       // theta -> (b, i, 32)
#pragma unroll
      for (int r = 0; r < 8; ++r) {
        int ch = mt * 16 + r + hh * 8;
        th[((size_t)(bb * NPIX + i0 + l15)) * DQ + ch] =
            (_Float16)(acc[r] + bcat[ch]);
      }
    } else {            // phi/g -> convpg (b, 160, 4096)
#pragma unroll
      for (int r = 0; r < 8; ++r) {
        int ch = mt * 16 + r + hh * 8;          // 32..191
        int row = ch - 32;                      // 0..159
        convpg[((size_t)(bb * CPG + row)) * HW_ + i0 + l15] =
            (_Float16)(acc[r] + bcat[ch]);
      }
    }
  }
}

// 2x2 maxpool of convpg rows 0..31 (phi) -> ph (b, M, 32)
__global__ void k_pool_phi(const _Float16* __restrict__ convpg, _Float16* __restrict__ ph) {
  int gid = blockIdx.x * 256 + threadIdx.x;     // b*1024 + m
  int bb = gid >> 10, m = gid & 1023;
  int my = m >> 5, mx = m & 31;
  int i00 = (my * 2) * 64 + mx * 2;
#pragma unroll 4
  for (int d = 0; d < DQ; ++d) {
    const _Float16* p = convpg + ((size_t)(bb * CPG + d)) * HW_ + i00;
    float v = fmaxf(fmaxf((float)p[0], (float)p[1]),
                    fmaxf((float)p[64], (float)p[65]));
    ph[(size_t)gid * DQ + d] = (_Float16)v;
  }
}

// 2x2 maxpool of convpg rows 32..159 (g) -> g16 (b, 128, M)
__global__ void k_pool_g(const _Float16* __restrict__ convpg, _Float16* __restrict__ g16) {
  int dvq = blockIdx.y;                         // 0..3
  int gid = blockIdx.x * 256 + threadIdx.x;     // b*1024 + m
  int bb = gid >> 10, m = gid & 1023;
  int my = m >> 5, mx = m & 31;
  int i00 = (my * 2) * 64 + mx * 2;
#pragma unroll 4
  for (int r = 0; r < 32; ++r) {
    int row = 32 + dvq * 32 + r;
    const _Float16* p = convpg + ((size_t)(bb * CPG + row)) * HW_ + i00;
    float v = fmaxf(fmaxf((float)p[0], (float)p[1]),
                    fmaxf((float)p[64], (float)p[65]));
    g16[((size_t)(bb * DV + dvq * 32 + r)) * MPOOL + m] = (_Float16)v;
  }
}

// ===========================================================================
//      Fallback scalar prep kernels (used when ws_size is small)
// ===========================================================================
__global__ void k_conv_theta(const float* __restrict__ x, const float* __restrict__ w,
                             const float* __restrict__ bias, _Float16* __restrict__ th) {
  __shared__ float ws[DQ * C_];
  int tid = threadIdx.x;
  for (int k = tid; k < DQ * C_; k += 256) ws[k] = w[k];
  __syncthreads();
  int gid = blockIdx.x * 256 + tid;
  int bb = gid >> 12, i = gid & 4095;
  const float* xb = x + (size_t)bb * C_ * HW_ + i;
  float acc[DQ];
#pragma unroll
  for (int d = 0; d < DQ; ++d) acc[d] = 0.f;
  for (int c = 0; c < C_; ++c) {
    float xv = xb[(size_t)c * HW_];
#pragma unroll
    for (int d = 0; d < DQ; ++d) acc[d] = fmaf(ws[d * C_ + c], xv, acc[d]);
  }
  _Float16* o = th + (size_t)gid * DQ;
#pragma unroll
  for (int d = 0; d < DQ; ++d) o[d] = (_Float16)(acc[d] + bias[d]);
}

__global__ void k_conv_pool_phi(const float* __restrict__ x, const float* __restrict__ w,
                                const float* __restrict__ bias, _Float16* __restrict__ ph) {
  __shared__ float ws[DQ * C_];
  int tid = threadIdx.x;
  for (int k = tid; k < DQ * C_; k += 256) ws[k] = w[k];
  __syncthreads();
  int gid = blockIdx.x * 256 + tid;
  int bb = gid >> 10, m = gid & 1023;
  int my = m >> 5, mx = m & 31;
  float best[DQ];
#pragma unroll
  for (int d = 0; d < DQ; ++d) best[d] = -3.0e38f;
  for (int dy = 0; dy < 2; ++dy)
    for (int dx = 0; dx < 2; ++dx) {
      int i = (2 * my + dy) * 64 + 2 * mx + dx;
      const float* xb = x + (size_t)bb * C_ * HW_ + i;
      float acc[DQ];
#pragma unroll
      for (int d = 0; d < DQ; ++d) acc[d] = 0.f;
      for (int c = 0; c < C_; ++c) {
        float xv = xb[(size_t)c * HW_];
#pragma unroll
        for (int d = 0; d < DQ; ++d) acc[d] = fmaf(ws[d * C_ + c], xv, acc[d]);
      }
#pragma unroll
      for (int d = 0; d < DQ; ++d) best[d] = fmaxf(best[d], acc[d]);
    }
  _Float16* o = ph + (size_t)gid * DQ;
#pragma unroll
  for (int d = 0; d < DQ; ++d) o[d] = (_Float16)(best[d] + bias[d]);
}

__global__ void k_conv_pool_g(const float* __restrict__ x, const float* __restrict__ w,
                              const float* __restrict__ bias, _Float16* __restrict__ g16) {
  __shared__ float ws[32 * C_];
  int tid = threadIdx.x;
  int dvq = blockIdx.y;
  for (int k = tid; k < 32 * C_; k += 256) ws[k] = w[dvq * 32 * C_ + k];
  __syncthreads();
  int gid = blockIdx.x * 256 + tid;
  int bb = gid >> 10, m = gid & 1023;
  int my = m >> 5, mx = m & 31;
  float best[32];
#pragma unroll
  for (int r = 0; r < 32; ++r) best[r] = -3.0e38f;
  for (int dy = 0; dy < 2; ++dy)
    for (int dx = 0; dx < 2; ++dx) {
      int i = (2 * my + dy) * 64 + 2 * mx + dx;
      const float* xb = x + (size_t)bb * C_ * HW_ + i;
      float acc[32];
#pragma unroll
      for (int r = 0; r < 32; ++r) acc[r] = 0.f;
      for (int c = 0; c < C_; ++c) {
        float xv = xb[(size_t)c * HW_];
#pragma unroll
        for (int r = 0; r < 32; ++r) acc[r] = fmaf(ws[r * C_ + c], xv, acc[r]);
      }
#pragma unroll
      for (int r = 0; r < 32; ++r) best[r] = fmaxf(best[r], acc[r]);
    }
#pragma unroll
  for (int r = 0; r < 32; ++r) {
    int dv = dvq * 32 + r;
    g16[((size_t)(bb * DV + dv)) * MPOOL + m] = (_Float16)(best[r] + bias[dv]);
  }
}

// ===========================================================================
// Pack out_w (256x128 f32) into WMMA A-fragment order.
// ===========================================================================
__global__ void k_pack_ow(const float* __restrict__ ow, _Float16* __restrict__ owf) {
  int idx = blockIdx.x * 256 + threadIdx.x;   // 32768 total
  int j    = idx & 15;
  int lane = (idx >> 4) & 31;
  int tq   = idx >> 9;                        // 0..63
  int q = tq & 3, ct = tq >> 2;
  int ch = ct * 16 + (lane & 15);
  int k  = ((j >> 3) << 4) + ((lane >> 4) << 3) + (j & 7);
  owf[idx] = (_Float16)ow[ch * DV + q * 32 + k];
}

// ===========================================================================
// Fused attention: one wave32 per 16-query tile (see round-1 notes).
// Cross-lane row-max reduction hoisted out of the energy loop.
// ===========================================================================
__global__ void __launch_bounds__(32)
k_attn(const _Float16* __restrict__ th, const _Float16* __restrict__ ph,
       const _Float16* __restrict__ g16, const _Float16* __restrict__ owf,
       const float* __restrict__ out_b, const float* __restrict__ gamma,
       const float* __restrict__ x, float* __restrict__ out) {
  __shared__ __align__(16) _Float16 e_lds[16 * MPOOL];  // 32 KB
  __shared__ float rmax_lds[16];
  __shared__ float rs_lds[16];

  const int lane = threadIdx.x;
  const int l15 = lane & 15;
  const int hh  = lane >> 4;
  const int wid = blockIdx.x;           // 0..2047
  const int bb  = wid >> 8;
  const int i0  = (wid & 255) << 4;

  // ---- A fragment: theta tile (16 queries x 32 d)
  size_t tb = ((size_t)(bb * NPIX + i0 + l15)) * DQ + hh * 8;
  v8h t0 = *(const v8h*)(th + tb);
  v8h t1 = *(const v8h*)(th + tb + 16);
  v16h a;
#pragma unroll
  for (int j = 0; j < 8; ++j) { a[j] = t0[j]; a[j + 8] = t1[j]; }

  float rmax[8];
#pragma unroll
  for (int r = 0; r < 8; ++r) rmax[r] = -3.0e38f;

  // ---- pass 1: energies (staged f16 in LDS) + per-lane running row max
  for (int jc = 0; jc < 64; ++jc) {
    int j0 = jc * 16;
    size_t pb = ((size_t)(bb * MPOOL + j0 + l15)) * DQ + hh * 8;
    v8h p0 = *(const v8h*)(ph + pb);
    v8h p1 = *(const v8h*)(ph + pb + 16);
    v16h bt;
#pragma unroll
    for (int j = 0; j < 8; ++j) { bt[j] = p0[j]; bt[j + 8] = p1[j]; }
    v8f c;
#pragma unroll
    for (int r = 0; r < 8; ++r) c[r] = 0.f;
    c = __builtin_amdgcn_wmma_f32_16x16x32_f16(false, a, false, bt, (short)0, c,
                                               false, false);
#pragma unroll
    for (int r = 0; r < 8; ++r) {
      e_lds[(r + hh * 8) * MPOOL + j0 + l15] = (_Float16)c[r];
      rmax[r] = fmaxf(rmax[r], c[r]);
    }
  }
  // single cross-lane reduction (within 16-lane halves) after the loop
#pragma unroll
  for (int r = 0; r < 8; ++r) {
    float vv = rmax[r];
    vv = fmaxf(vv, __shfl_xor(vv, 1, 32));
    vv = fmaxf(vv, __shfl_xor(vv, 2, 32));
    vv = fmaxf(vv, __shfl_xor(vv, 4, 32));
    vv = fmaxf(vv, __shfl_xor(vv, 8, 32));
    rmax[r] = vv;
  }
  __syncthreads();
  if (l15 == 0) {
#pragma unroll
    for (int r = 0; r < 8; ++r) rmax_lds[hh * 8 + r] = rmax[r];
  }
  __syncthreads();
  const float rm = rmax_lds[l15];   // max of the row this lane owns in A layout

  // ---- pass 2: P = exp(e - rm); O += P x g^T
  v8f o[8];
#pragma unroll
  for (int t = 0; t < 8; ++t)
#pragma unroll
    for (int r = 0; r < 8; ++r) o[t][r] = 0.f;
  float rsum = 0.f;

  for (int kc = 0; kc < 32; ++kc) {
    int kc0 = kc * 32;
    const _Float16* eb = e_lds + l15 * MPOOL + kc0 + hh * 8;
    v8h e0 = *(const v8h*)eb;          // k = hh*8 + j
    v8h e1 = *(const v8h*)(eb + 16);   // k = 16 + hh*8 + j
    v16h p;
#pragma unroll
    for (int j = 0; j < 8; ++j) {
      float pe0 = __expf((float)e0[j] - rm); rsum += pe0; p[j]     = (_Float16)pe0;
      float pe1 = __expf((float)e1[j] - rm); rsum += pe1; p[j + 8] = (_Float16)pe1;
    }
#pragma unroll
    for (int t = 0; t < 8; ++t) {
      size_t gb = ((size_t)(bb * DV + t * 16 + l15)) * MPOOL + kc0 + hh * 8;
      v8h g0 = *(const v8h*)(g16 + gb);
      v8h g1 = *(const v8h*)(g16 + gb + 16);
      v16h bt;
#pragma unroll
      for (int j = 0; j < 8; ++j) { bt[j] = g0[j]; bt[j + 8] = g1[j]; }
      o[t] = __builtin_amdgcn_wmma_f32_16x16x32_f16(false, p, false, bt, (short)0,
                                                    o[t], false, false);
    }
  }

  // ---- normalize by row sums (lanes m and m+16 hold halves of row m)
  float tot = rsum + __shfl_xor(rsum, 16, 32);
  if (hh == 0) rs_lds[l15] = tot;
  __syncthreads();
  float inv[8];
#pragma unroll
  for (int r = 0; r < 8; ++r) inv[r] = 1.0f / rs_lds[r + hh * 8];
#pragma unroll
  for (int t = 0; t < 8; ++t)
#pragma unroll
    for (int r = 0; r < 8; ++r) o[t][r] *= inv[r];

  // ---- stage O (16 x 128) in LDS (reuse e_lds) for B-fragment reads
  __syncthreads();
  _Float16* o16 = e_lds;
#pragma unroll
  for (int t = 0; t < 8; ++t)
#pragma unroll
    for (int r = 0; r < 8; ++r)
      o16[(r + hh * 8) * DV + t * 16 + l15] = (_Float16)o[t][r];
  __syncthreads();

  // ---- final conv (256x128)x(128x16) + fused gamma/bias/residual
  const float gam = gamma[0];
  for (int ct = 0; ct < 16; ++ct) {
    v8f acc;
#pragma unroll
    for (int r = 0; r < 8; ++r) acc[r] = 0.f;
#pragma unroll
    for (int q = 0; q < 4; ++q) {
      const _Float16* wb = owf + ((size_t)((ct * 4 + q) * 32 + lane)) * 16;
      v8h w0 = *(const v8h*)wb;
      v8h w1 = *(const v8h*)(wb + 8);
      v16h aw;
#pragma unroll
      for (int j = 0; j < 8; ++j) { aw[j] = w0[j]; aw[j + 8] = w1[j]; }
      const _Float16* ob = o16 + l15 * DV + q * 32 + hh * 8;
      v8h b0 = *(const v8h*)ob;
      v8h b1 = *(const v8h*)(ob + 16);
      v16h bo;
#pragma unroll
      for (int j = 0; j < 8; ++j) { bo[j] = b0[j]; bo[j + 8] = b1[j]; }
      acc = __builtin_amdgcn_wmma_f32_16x16x32_f16(false, aw, false, bo, (short)0,
                                                   acc, false, false);
    }
#pragma unroll
    for (int r = 0; r < 8; ++r) {
      int cch = ct * 16 + r + hh * 8;
      size_t idx = ((size_t)(bb * C_ + cch)) * HW_ + i0 + l15;
      out[idx] = gam * (acc[r] + out_b[cch]) + x[idx];
    }
  }
}

// ===========================================================================
extern "C" void kernel_launch(void* const* d_in, const int* in_sizes, int n_in,
                              void* d_out, int out_size, void* d_ws, size_t ws_size,
                              hipStream_t stream) {
  const float* x       = (const float*)d_in[0];
  const float* theta_w = (const float*)d_in[1];
  const float* theta_b = (const float*)d_in[2];
  const float* phi_w   = (const float*)d_in[3];
  const float* phi_b   = (const float*)d_in[4];
  const float* g_w     = (const float*)d_in[5];
  const float* g_b     = (const float*)d_in[6];
  const float* out_w   = (const float*)d_in[7];
  const float* out_b   = (const float*)d_in[8];
  const float* gamma   = (const float*)d_in[9];
  float* out = (float*)d_out;

  // Workspace layout (every region fully rewritten each call):
  //   th     @ 0        : 2 MB   (b, N, 32)  f16
  //   ph     @ 2097152  : 512 KB (b, M, 32)  f16
  //   g16    @ 2621440  : 2 MB   (b,128, M)  f16
  //   owf    @ 4718592  : 64 KB  packed out_w fragments
  //   -- WMMA-prep extras --
  //   wpk    @ 4784128  : 96 KB  packed theta/phi/g weight fragments
  //   bcat   @ 4882432  : 1 KB   concatenated biases f32
  //   xt     @ 4883456  : 16 MB  (b, N, 256) f16
  //   convpg @ 21660672 : 10 MB  (b, 160, N) f16 full-res phi/g conv
  char* wsb = (char*)d_ws;
  _Float16* th     = (_Float16*)(wsb);
  _Float16* ph     = (_Float16*)(wsb + 2097152);
  _Float16* g16    = (_Float16*)(wsb + 2621440);
  _Float16* owf    = (_Float16*)(wsb + 4718592);
  _Float16* wpk    = (_Float16*)(wsb + 4784128);
  float*    bcat   = (float*)   (wsb + 4882432);
  _Float16* xt     = (_Float16*)(wsb + 4883456);
  _Float16* convpg = (_Float16*)(wsb + 21660672);

  k_pack_ow<<<128, 256, 0, stream>>>(out_w, owf);

  if (ws_size >= 32146432ull) {
    // WMMA prep path: transpose x, pack weights, fused conv, pool.
    k_transpose_x<<<128, 256, 0, stream>>>(x, xt);
    k_pack_wcat<<<192, 256, 0, stream>>>(theta_w, phi_w, g_w,
                                         theta_b, phi_b, g_b, wpk, bcat);
    k_conv_wmma<<<2048, 32, 0, stream>>>(xt, wpk, bcat, th, convpg);
    k_pool_phi<<<32, 256, 0, stream>>>(convpg, ph);
    k_pool_g<<<dim3(32, 4), 256, 0, stream>>>(convpg, g16);
  } else {
    // Fallback scalar prep (small workspace).
    k_conv_theta<<<128, 256, 0, stream>>>(x, theta_w, theta_b, th);
    k_conv_pool_phi<<<32, 256, 0, stream>>>(x, phi_w, phi_b, ph);
    k_conv_pool_g<<<dim3(32, 4), 256, 0, stream>>>(x, g_w, g_b, g16);
  }

  k_attn<<<2048, 32, 0, stream>>>(th, ph, g16, owf, out_b, gamma, x, out);
}
